// OctreeConv_59072980189440
// MI455X (gfx1250) — matrix operand order repaired and checked by reading, not compiled, not run on hardware
//
#include <hip/hip_runtime.h>
#include <stdint.h>

typedef __attribute__((ext_vector_type(16))) _Float16 v16h;
typedef __attribute__((ext_vector_type(8)))  _Float16 v8h;
typedef __attribute__((ext_vector_type(4)))  _Float16 v4h;
typedef __attribute__((ext_vector_type(8)))  float    v8f;
typedef __attribute__((ext_vector_type(4)))  float    v4f;

#define N_OCT   262144
#define C_IN    32
#define C_OUT   64
#define K3      27
// Weight fragment blob: [k][cg][lane][16 f16]  (cg = 16-col group of C_OUT)
#define WBLOB_HALFS (K3 * 4 * 32 * 16)          // 55296 halfs
#define WBLOB_BYTES (WBLOB_HALFS * 2)           // 110592 bytes
// f16 input copy has one extra all-zero row at index N_OCT (invalid-neighbor target)
#define IN_HALFS ((N_OCT + 1) * C_IN)

// ---------------- Prepass 1: fp32 -> fp16 conversion + zero pad row -------
__global__ __launch_bounds__(256) void cvt_input_f16(const float* __restrict__ src,
                                                     _Float16* __restrict__ dst) {
  int i = (blockIdx.x * 256 + threadIdx.x) * 4;
  if (i >= IN_HALFS) return;
  v4h h = {};
  if (i < N_OCT * C_IN) {
    v4f v = *(const v4f*)(src + i);
    h.x = (_Float16)v.x; h.y = (_Float16)v.y; h.z = (_Float16)v.z; h.w = (_Float16)v.w;
  }
  *(v4h*)(dst + i) = h;
}

// ---------------- Prepass 2: pack weights into per-lane B fragments -------
// B (32x16 f16) layout used here (contiguous K, same as A): lane L -> column n = L&15; K base kb = (L<16)?0:16;
// VGPR v holds K=kb+2v (lo16) and kb+2v+1 (hi16)  => lane's 16 halfs are K=kb..kb+15.
__global__ __launch_bounds__(256) void pack_weights(const float* __restrict__ W,
                                                    _Float16* __restrict__ blob) {
  int t = blockIdx.x * 256 + threadIdx.x;     // 0 .. K3*4*32-1
  if (t >= K3 * 4 * 32) return;
  int lane = t & 31;
  int cg   = (t >> 5) & 3;
  int k    = t >> 7;
  int n    = lane & 15;
  int kb   = (lane < 16) ? 0 : 16;
  _Float16* dst = blob + (size_t)t * 16;
  const float* src = W + (size_t)k * (C_IN * C_OUT) + (size_t)kb * C_OUT + (cg * 16 + n);
#pragma unroll
  for (int j = 0; j < 16; ++j)
    dst[j] = (_Float16)src[j * C_OUT];
}

// Clamp neighbor index: -1 -> 0xFFFFFFFF -> min -> N_OCT (the zero row).
static __device__ __forceinline__ uint32_t safe_row(int i) {
  uint32_t u = (uint32_t)i;
  return u > (uint32_t)N_OCT ? (uint32_t)N_OCT : u;   // v_min_u32
}

// ---------------- Main kernel: pipelined gather + WMMA GEMM ---------------
// Each wave owns a 32-row x 64-col output tile: 2 A fragments, 8 accumulators.
__global__ __launch_bounds__(256) void octree_conv_wmma(
    const _Float16* __restrict__ inF16,   // [N+1][32] f16 (last row zeros)
    const _Float16* __restrict__ wblob,   // WBLOB_BYTES packed B fragments
    const int*      __restrict__ neigh,   // [N][27]
    float*          __restrict__ out) {   // [N][64]
  __shared__ __align__(32) _Float16 ldsW[WBLOB_HALFS];

  const int tid = threadIdx.x;

  // ---- Stage weight blob into LDS via CDNA5 async global->LDS copies ----
  {
    uint32_t ldsbase = (uint32_t)(uintptr_t)(&ldsW[0]);
    uint64_t gbase   = (uint64_t)(uintptr_t)wblob;
#pragma unroll 1
    for (int r = 0; r < 27; ++r) {               // 27 * 256 lanes * 16B = 110592 B
      uint32_t off     = (uint32_t)r * 4096u + (uint32_t)tid * 16u;
      uint32_t ldsaddr = ldsbase + off;
      asm volatile("global_load_async_to_lds_b128 %0, %1, %2"
                   :: "v"(ldsaddr), "v"(off), "s"(gbase)
                   : "memory");
    }
    asm volatile("s_wait_asynccnt 0" ::: "memory");
  }
  __syncthreads();

  const int wave = tid >> 5;
  const int lane = tid & 31;
  const int tile = blockIdx.x * 8 + wave;   // 32-row output tile
  const int n0   = tile * 32;
  const int m    = lane & 15;               // A-fragment row = lane&15
  const int hi   = lane >> 4;               // half-wave select
  const int kb2  = hi * 32;                 // byte offset of A K-chunk (K base * 2B)

  const int* nrowA = neigh + (size_t)(n0 + m) * K3;
  const int* nrowB = neigh + (size_t)(n0 + 16 + m) * K3;
  const char* ibase = (const char*)inF16;
  const v16h* wlds  = (const v16h*)ldsW;

  v8f c[8];
#pragma unroll
  for (int i = 0; i < 8; ++i) c[i] = (v8f){};

  // ---- Software pipeline: indices 2 taps ahead, A gathers 1 tap ahead ----
  int iA = nrowA[1], iB = nrowB[1];         // index for tap k+1
  v8h a0, a1, b0, b1;                       // current-tap fragments
  {
    const char* rpA = ibase + (size_t)safe_row(nrowA[0]) * (C_IN * 2);
    const char* rpB = ibase + (size_t)safe_row(nrowB[0]) * (C_IN * 2);
    a0 = *(const v8h*)(rpA + kb2);          // K = kb .. kb+7
    a1 = *(const v8h*)(rpA + kb2 + 16);     // K = kb+8 .. kb+15
    b0 = *(const v8h*)(rpB + kb2);
    b1 = *(const v8h*)(rpB + kb2 + 16);
  }

#pragma unroll 1
  for (int k = 0; k < K3; ++k) {
    // prefetch tap k+2 indices
    int inA = (k + 2 < K3) ? nrowA[k + 2] : 0;
    int inB = (k + 2 < K3) ? nrowB[k + 2] : 0;

    // issue tap k+1 gathers (invalid -> zero row, no masking needed)
    const char* rpA = ibase + (size_t)safe_row(iA) * (C_IN * 2);
    const char* rpB = ibase + (size_t)safe_row(iB) * (C_IN * 2);
    v8h na0 = *(const v8h*)(rpA + kb2);
    v8h na1 = *(const v8h*)(rpA + kb2 + 16);
    v8h nb0 = *(const v8h*)(rpB + kb2);
    v8h nb1 = *(const v8h*)(rpB + kb2 + 16);

    v16h A1 = __builtin_shufflevector(a0, a1, 0, 1, 2, 3, 4, 5, 6, 7,
                                      8, 9, 10, 11, 12, 13, 14, 15);
    v16h A2 = __builtin_shufflevector(b0, b1, 0, 1, 2, 3, 4, 5, 6, 7,
                                      8, 9, 10, 11, 12, 13, 14, 15);

    // B fragments from LDS; each reused by two WMMAs (two row-halves)
    v16h Bf;
    Bf = wlds[(k * 4 + 0) * 32 + lane];
    c[0] = __builtin_amdgcn_wmma_f32_16x16x32_f16(false, A1, false, Bf, (short)0, c[0], false, false);
    c[4] = __builtin_amdgcn_wmma_f32_16x16x32_f16(false, A2, false, Bf, (short)0, c[4], false, false);
    Bf = wlds[(k * 4 + 1) * 32 + lane];
    c[1] = __builtin_amdgcn_wmma_f32_16x16x32_f16(false, A1, false, Bf, (short)0, c[1], false, false);
    c[5] = __builtin_amdgcn_wmma_f32_16x16x32_f16(false, A2, false, Bf, (short)0, c[5], false, false);
    Bf = wlds[(k * 4 + 2) * 32 + lane];
    c[2] = __builtin_amdgcn_wmma_f32_16x16x32_f16(false, A1, false, Bf, (short)0, c[2], false, false);
    c[6] = __builtin_amdgcn_wmma_f32_16x16x32_f16(false, A2, false, Bf, (short)0, c[6], false, false);
    Bf = wlds[(k * 4 + 3) * 32 + lane];
    c[3] = __builtin_amdgcn_wmma_f32_16x16x32_f16(false, A1, false, Bf, (short)0, c[3], false, false);
    c[7] = __builtin_amdgcn_wmma_f32_16x16x32_f16(false, A2, false, Bf, (short)0, c[7], false, false);

    // rotate pipeline
    a0 = na0; a1 = na1; b0 = nb0; b1 = nb1;
    iA = inA; iB = inB;
  }

  // C/D layout (16x16 f32): VGPR r, lane L -> row = r + 8*hi, col = (L&15) + 16*cg
  float* oA = out + (size_t)(n0 + hi * 8) * C_OUT + m;
  float* oB = out + (size_t)(n0 + 16 + hi * 8) * C_OUT + m;
#pragma unroll
  for (int r = 0; r < 8; ++r) {
    oA[r * C_OUT + 0]  = c[0][r];
    oA[r * C_OUT + 16] = c[1][r];
    oA[r * C_OUT + 32] = c[2][r];
    oA[r * C_OUT + 48] = c[3][r];
    oB[r * C_OUT + 0]  = c[4][r];
    oB[r * C_OUT + 16] = c[5][r];
    oB[r * C_OUT + 32] = c[6][r];
    oB[r * C_OUT + 48] = c[7][r];
  }
}

// --------------------------------------------------------------------------
extern "C" void kernel_launch(void* const* d_in, const int* in_sizes, int n_in,
                              void* d_out, int out_size, void* d_ws, size_t ws_size,
                              hipStream_t stream) {
  const float* inputsignal = (const float*)d_in[0];   // [N][32] f32
  const float* weights     = (const float*)d_in[1];   // [27][32][64] f32
  const int*   neigh       = (const int*)d_in[2];     // [N][27] i32
  float*       out         = (float*)d_out;           // [N][64] f32

  _Float16* inF16 = (_Float16*)d_ws;                          // (N+1)*32 halfs
  _Float16* wblob = inF16 + (size_t)IN_HALFS;                 // +110592 B

  const int cvt_threads = (IN_HALFS + 3) / 4;         // 4 halfs per thread
  cvt_input_f16<<<(cvt_threads + 255) / 256, 256, 0, stream>>>(inputsignal, inF16);
  pack_weights<<<(K3 * 4 * 32 + 255) / 256, 256, 0, stream>>>(weights, wblob);

  const int tiles  = N_OCT / 32;   // 8192 tiles of 32 rows
  const int blocks = tiles / 8;    // 1024 (8 waves per 256-thread block)
  octree_conv_wmma<<<blocks, 256, 0, stream>>>(inF16, wblob, neigh, out);
}
